// WaveletKAN_5841155522678
// MI455X (gfx1250) — matrix operand order repaired and checked by reading, not compiled, not run on hardware
//
#include <hip/hip_runtime.h>
#include <math.h>

typedef __attribute__((ext_vector_type(16))) _Float16 v16h;
typedef __attribute__((ext_vector_type(8)))  float    v8f;
typedef __attribute__((ext_vector_type(4)))  int      v4i;

#define IN_DIM  4096
#define OUT_DIM 4096
#define BATCH   128
#define S0      64
#define NLEV    4
#define PRUNE_Q 0.4f

// ---- CDNA5 async LDS<->global DMA (guarded: falls back to plain copies) ----
#if defined(__has_builtin)
#if __has_builtin(__builtin_amdgcn_global_load_async_to_lds_b128)
#define HAVE_ASYNC_LD 1
#endif
#if __has_builtin(__builtin_amdgcn_global_store_async_from_lds_b128)
#define HAVE_ASYNC_ST 1
#endif
#endif

// Builtin expects (v4i addrspace(1)*, v4i addrspace(3)*, imm, imm) per the
// clang diagnostic ("int __device__ *" == addrspace(1) int4 pointer in HIP).
#define GPTR(p) ((__attribute__((address_space(1))) v4i*)(p))
#define LPTR(p) ((__attribute__((address_space(3))) v4i*)(p))

// ---------------------------------------------------------------------------
// Wavelet branch: one workgroup per batch image. out = wavelet_out + bias
// ---------------------------------------------------------------------------

__device__ float kth_of_abs(const float* p, int n, int k) {
    // k-th smallest (0-based) of |p[0..n)| via rank counting (n <= 32)
    for (int i = 0; i < n; ++i) {
        float v = fabsf(p[i]);
        int less = 0, eq = 0;
        for (int j = 0; j < n; ++j) {
            float u = fabsf(p[j]);
            less += (u < v);
            eq   += (u == v);
        }
        if (less <= k && k < less + eq) return v;
    }
    return 0.0f;
}

__device__ float quantile_abs(const float* p, int n, float q) {
    float pos = q * (float)(n - 1);
    int   lo  = (int)floorf(pos);
    float f   = pos - (float)lo;
    float vlo = kth_of_abs(p, n, lo);
    float vhi = (f > 0.0f) ? kth_of_abs(p, n, lo + 1) : vlo;
    return vlo + (vhi - vlo) * f;
}

__global__ __launch_bounds__(256)
void wavelet_kernel(const float* __restrict__ x,
                    const float* __restrict__ bias,
                    const float* __restrict__ sf,   // (4,4) scale factors
                    float* __restrict__ out) {
    __shared__ float cur[S0 * S0];          // 16 KB: cA pyramid / recon ping
    __shared__ float nxt[S0 * S0];          // 16 KB: cA temp     / recon pong
    __shared__ float bH[1360];              // 32^2+16^2+8^2+4^2 per band
    __shared__ float bV[1360];
    __shared__ float bD[1360];

    const int b   = blockIdx.x;
    const int tid = threadIdx.x;

    // ---- image -> LDS (async DMA path if available) ----
#ifdef HAVE_ASYNC_LD
    for (int i = tid * 4; i < S0 * S0; i += 256 * 4)
        __builtin_amdgcn_global_load_async_to_lds_b128(
            GPTR(x + b * IN_DIM + i), LPTR(&cur[i]), 0, 0);
    asm volatile("s_wait_asynccnt 0x0" ::: "memory");
#else
    for (int i = tid; i < S0 * S0; i += 256) cur[i] = x[b * IN_DIM + i];
#endif
    __syncthreads();

    const int loff[NLEV] = {0, 1024, 1280, 1344};

    // ---------------- forward DWT + per-row quantile prune ----------------
    int S = S0;
    for (int l = 0; l < NLEV; ++l) {
        const int So = S >> 1;
        const float s0 = sf[l * 4 + 0], s1 = sf[l * 4 + 1];
        const float s2 = sf[l * 4 + 2], s3 = sf[l * 4 + 3];

        for (int idx = tid; idx < So * So; idx += 256) {
            int i = idx / So, j = idx % So;
            float a  = cur[(2 * i)     * S + 2 * j];
            float bb = cur[(2 * i)     * S + 2 * j + 1];
            float c  = cur[(2 * i + 1) * S + 2 * j];
            float d  = cur[(2 * i + 1) * S + 2 * j + 1];
            nxt[idx]            = (a + bb + c + d) * 0.5f * s0;
            bH[loff[l] + idx]   = (a - bb + c - d) * 0.5f * s1;
            bV[loff[l] + idx]   = (a + bb - c - d) * 0.5f * s2;
            bD[loff[l] + idx]   = (a - bb - c + d) * 0.5f * s3;
        }
        __syncthreads();

        // threshold & prune: one thread per (band,row)
        for (int r = tid; r < 4 * So; r += 256) {
            int band = r / So, row = r % So;
            float* p;
            if      (band == 0) p = &nxt[row * So];
            else if (band == 1) p = &bH[loff[l] + row * So];
            else if (band == 2) p = &bV[loff[l] + row * So];
            else                p = &bD[loff[l] + row * So];
            float thr = quantile_abs(p, So, PRUNE_Q);
            for (int j = 0; j < So; ++j) {
                float v = p[j];
                p[j] = (fabsf(v) > thr) ? v : 0.0f;
            }
        }
        __syncthreads();

        for (int idx = tid; idx < So * So; idx += 256) cur[idx] = nxt[idx];
        __syncthreads();
        S = So;
    }

    // ---------------- inverse DWT (even # of swaps -> ends in cur) --------
    float* rec = cur;   // holds pruned level-3 cA (4x4) in [0,16)
    float* dst = nxt;
    int Si = S0 >> NLEV;  // 4
    for (int l = NLEV - 1; l >= 0; --l) {
        for (int idx = tid; idx < Si * Si; idx += 256) {
            int i = idx / Si, j = idx % Si;
            float cA = rec[idx];
            float cH = bH[loff[l] + idx];
            float cV = bV[loff[l] + idx];
            float cD = bD[loff[l] + idx];
            float a  = (cA + cH + cV + cD) * 0.5f;
            float bb = (cA - cH + cV - cD) * 0.5f;
            float c  = (cA + cH - cV - cD) * 0.5f;
            float d  = (cA - cH - cV + cD) * 0.5f;
            int W = 2 * Si;
            dst[(2 * i)     * W + 2 * j]     = a;
            dst[(2 * i)     * W + 2 * j + 1] = bb;
            dst[(2 * i + 1) * W + 2 * j]     = c;
            dst[(2 * i + 1) * W + 2 * j + 1] = d;
        }
        __syncthreads();
        float* t = rec; rec = dst; dst = t;
        Si <<= 1;
    }
    // NLEV == 4 swaps: reconstruction is in `cur`.

    // ---- add bias in LDS, then LDS -> global (async DMA path) ----
    for (int i = tid; i < IN_DIM; i += 256) cur[i] += bias[i];
    __syncthreads();
#ifdef HAVE_ASYNC_ST
    for (int i = tid * 4; i < IN_DIM; i += 256 * 4)
        __builtin_amdgcn_global_store_async_from_lds_b128(
            GPTR(out + b * OUT_DIM + i), LPTR(&cur[i]), 0, 0);
    asm volatile("s_wait_asynccnt 0x0" ::: "memory");
#else
    for (int i = tid; i < IN_DIM; i += 256)
        out[b * OUT_DIM + i] = cur[i];
#endif
}

// ---------------------------------------------------------------------------
// GEMM: out += x @ W^T  (M=128, N=4096, K=4096), fp32 -> f16 WMMA, f32 acc.
// Grid: 64 WGs (N tiles of 64). Each WG: 8 waves = full M=128, weights read
// exactly once (64 MB stream = the roofline term, ~2.8us at 23.3 TB/s).
// ---------------------------------------------------------------------------

#define TN 64
#define TK 32

union FragA { v16h v; _Float16 h[16]; };
union FragB { v16h v; _Float16 h[16]; };
union AccU  { v8f  v; float    f[8];  };

__global__ __launch_bounds__(256)
void gemm_kernel(const float* __restrict__ x,
                 const float* __restrict__ w,
                 float* __restrict__ out) {
    __shared__ _Float16 As[BATCH * TK];  // 128 x 32 f16 = 8 KB
    __shared__ _Float16 Bs[TN * TK];     //  64 x 32 f16 = 4 KB

    const int tid  = threadIdx.x;
    const int wave = tid >> 5;
    const int lane = tid & 31;
    const int n0   = blockIdx.x * TN;

    v8f acc[4] = {v8f{0}, v8f{0}, v8f{0}, v8f{0}};

    const int arow  = wave * 16 + (lane & 15);       // M row for A fragment
    const int kbA   = (lane & 16) ? 16 : 0;          // A: K half base
    const int kbB   = (lane & 16) ? 16 : 0;          // B: K half base
    const int ncol  = lane & 15;                     // B: column within subtile

    for (int kt = 0; kt < IN_DIM; kt += TK) {
        // ---- stage A tile (x: 128 x 32 fp32 -> f16) ----
        #pragma unroll
        for (int r = 0; r < 4; ++r) {
            int slot = tid + 256 * r;                // 1024 float4 slots
            int row  = slot >> 3;
            int c4   = slot & 7;
            float4 v = *reinterpret_cast<const float4*>(x + row * IN_DIM + kt + c4 * 4);
            _Float16* p = &As[row * TK + c4 * 4];
            p[0] = (_Float16)v.x; p[1] = (_Float16)v.y;
            p[2] = (_Float16)v.z; p[3] = (_Float16)v.w;
        }
        // ---- stage B tile (W rows n0..n0+63 x 32 fp32 -> f16) ----
        #pragma unroll
        for (int r = 0; r < 2; ++r) {
            int slot = tid + 256 * r;                // 512 float4 slots
            int row  = slot >> 3;
            int c4   = slot & 7;
            float4 v = *reinterpret_cast<const float4*>(w + (n0 + row) * IN_DIM + kt + c4 * 4);
            _Float16* p = &Bs[row * TK + c4 * 4];
            p[0] = (_Float16)v.x; p[1] = (_Float16)v.y;
            p[2] = (_Float16)v.z; p[3] = (_Float16)v.w;
        }
        // prefetch next weight tile (global_prefetch_b8)
        if (kt + TK < IN_DIM)
            __builtin_prefetch(w + (n0 + (tid >> 2)) * IN_DIM + kt + TK, 0, 0);
        __syncthreads();

        // ---- A fragment: 16x32 f16, contiguous k = 16*(lane/16) + i, same as B ----
        FragA fa;
        #pragma unroll
        for (int i = 0; i < 8; ++i) fa.h[i]     = As[arow * TK + kbA + i];
        #pragma unroll
        for (int i = 0; i < 8; ++i) fa.h[8 + i] = As[arow * TK + kbA + 8 + i];

        // ---- 4 N-subtiles of 16 per wave ----
        #pragma unroll
        for (int j = 0; j < 4; ++j) {
            FragB fb;
            const _Float16* bp = &Bs[(j * 16 + ncol) * TK + kbB];
            #pragma unroll
            for (int i = 0; i < 16; ++i) fb.h[i] = bp[i];
            acc[j] = __builtin_amdgcn_wmma_f32_16x16x32_f16(
                false, fa.v, false, fb.v, (short)0, acc[j], false, false);
        }
        __syncthreads();
    }

    // ---- epilogue: out += acc (one owner per element, no atomics) ----
    const int mbase = wave * 16 + ((lane & 16) ? 8 : 0);
    #pragma unroll
    for (int j = 0; j < 4; ++j) {
        AccU u; u.v = acc[j];
        const int n = n0 + j * 16 + (lane & 15);
        #pragma unroll
        for (int r = 0; r < 8; ++r) {
            const int m = mbase + r;
            out[m * OUT_DIM + n] += u.f[r];
        }
    }
}

// ---------------------------------------------------------------------------

extern "C" void kernel_launch(void* const* d_in, const int* in_sizes, int n_in,
                              void* d_out, int out_size, void* d_ws, size_t ws_size,
                              hipStream_t stream) {
    const float* x    = (const float*)d_in[0];
    const float* w    = (const float*)d_in[1];
    const float* bias = (const float*)d_in[2];
    const float* sf   = (const float*)d_in[3];
    float* out        = (float*)d_out;

    // 1) out = wavelet_out + bias   (writes every element)
    wavelet_kernel<<<BATCH, 256, 0, stream>>>(x, bias, sf, out);
    // 2) out += x @ W^T             (stream-ordered after kernel 1)
    gemm_kernel<<<OUT_DIM / TN, 256, 0, stream>>>(x, w, out);
}
